// SelfNormAttention_5660766896671
// MI455X (gfx1250) — matrix operand order repaired and checked by reading, not compiled, not run on hardware
//
#include <hip/hip_runtime.h>

typedef __attribute__((ext_vector_type(16))) _Float16 v16h;
typedef __attribute__((ext_vector_type(8)))  float    v8f;

namespace {
constexpr int kB = 4, kS = 2048, kH = 16, kD = 64;
constexpr int kBlockQ = 128;   // queries per block (8 waves x 16)
constexpr int kChunkS = 32;    // keys/values per iteration

union V16H { v16h v; _Float16 h[16]; uint4 q[2]; unsigned u[8]; };
union V8F  { v8f  v; float f[8]; };
union H8   { _Float16 h[8]; uint4 u; };
union H2   { _Float16 h[2]; unsigned u; };
}

__global__ __launch_bounds__(256)
void attn_fwd(const float* __restrict__ qk, const float* __restrict__ vv,
              const float* __restrict__ nn, float* __restrict__ out)
{
  // K chunk row-major [s][d], V chunk transposed [d][s], both f16
  __shared__ __align__(16) _Float16 ldsK[kChunkS * kD];
  __shared__ __align__(16) _Float16 ldsV[kD * kChunkS];

  const int tid  = threadIdx.x;
  const int lane = tid & 31;
  const int wave = tid >> 5;
  const int lh   = lane & 15;   // N index (query col) / M row for A operands
  const int cc   = lane >> 4;   // lane-group selector in WMMA layouts

  const int qt = blockIdx.x & 15;
  const int h  = (blockIdx.x >> 4) & (kH - 1);
  const int b  = blockIdx.x >> 8;

  const int t_base = qt * kBlockQ + wave * 16;
  const int t      = t_base + lh;        // this lane's query row (fixed!)
  const int t_max  = t_base + 15;

  const float scale = 1.0f / __expf(nn[(b * kS + t) * kH + h]);

  // ---- loop-invariant Q^T B-operands: B[k][n] = Q[t_base+n][32*dc + 16*cc + j]
  const size_t q_row = ((size_t)(b * kS + t)) * (2 * kH * kD) + (size_t)h * kD;
  V16H bq[2];
#pragma unroll
  for (int dc = 0; dc < 2; ++dc) {
    const float4* p = (const float4*)(qk + q_row + dc * 32 + cc * 16);
#pragma unroll
    for (int j = 0; j < 4; ++j) {
      float4 f = p[j];
      bq[dc].h[j * 4 + 0] = (_Float16)f.x;
      bq[dc].h[j * 4 + 1] = (_Float16)f.y;
      bq[dc].h[j * 4 + 2] = (_Float16)f.z;
      bq[dc].h[j * 4 + 3] = (_Float16)f.w;
    }
  }

  // O^T accumulators: 4 tiles of 16(d) x 16(t), fp32
  v8f acc[4];
  const v8f zero8 = {0.f, 0.f, 0.f, 0.f, 0.f, 0.f, 0.f, 0.f};
#pragma unroll
  for (int i = 0; i < 4; ++i) acc[i] = zero8;

  // ---- staging thread mapping (block-wide, coalesced) ----
  const int krow = tid >> 3;            // 0..31 : s within chunk
  const int kcol = (tid & 7) * 8;       // 8-float column run
  const int vd   = tid & 63;            // d column
  const int vsg  = tid >> 6;            // 0..3 : group of 8 s values
  const size_t kbase = (size_t)b * kS * 2 * kH * kD + (size_t)kH * kD
                     + (size_t)h * kD + kcol;
  const size_t vbase = (size_t)b * kS * kH * kD + (size_t)h * kD + vd;

  const int nchunks = qt * 4 + 4;       // causal: s <= t_base_block + 127

  float gk[8], gv[8];
  // prologue: load chunk 0 into registers
  {
    const float4* p = (const float4*)(qk + kbase + (size_t)krow * (2 * kH * kD));
    float4 f0 = p[0], f1 = p[1];
    gk[0] = f0.x; gk[1] = f0.y; gk[2] = f0.z; gk[3] = f0.w;
    gk[4] = f1.x; gk[5] = f1.y; gk[6] = f1.z; gk[7] = f1.w;
#pragma unroll
    for (int j = 0; j < 8; ++j)
      gv[j] = vv[vbase + (size_t)(vsg * 8 + j) * (kH * kD)];
  }

  for (int ch = 0; ch < nchunks; ++ch) {
    const int s0 = ch * kChunkS;

    // commit staged registers to LDS as f16
    {
      H8 hk, hv;
#pragma unroll
      for (int j = 0; j < 8; ++j) { hk.h[j] = (_Float16)gk[j]; hv.h[j] = (_Float16)gv[j]; }
      *(uint4*)&ldsK[krow * kD + kcol]       = hk.u;
      *(uint4*)&ldsV[vd * kChunkS + vsg * 8] = hv.u;
    }
    __syncthreads();

    // software pipeline: issue next chunk's global loads now
    if (ch + 1 < nchunks) {
      const int s1 = s0 + kChunkS;
      const float4* p = (const float4*)(qk + kbase + (size_t)(s1 + krow) * (2 * kH * kD));
      float4 f0 = p[0], f1 = p[1];
      gk[0] = f0.x; gk[1] = f0.y; gk[2] = f0.z; gk[3] = f0.w;
      gk[4] = f1.x; gk[5] = f1.y; gk[6] = f1.z; gk[7] = f1.w;
#pragma unroll
      for (int j = 0; j < 8; ++j)
        gv[j] = vv[vbase + (size_t)(s1 + vsg * 8 + j) * (kH * kD)];
    }

    if (s0 <= t_max) {  // wave-uniform causal skip
      // ---- S^T = K · Q^T : two 16(s) x 16(t) tiles ----
      v8f sc[2];
#pragma unroll
      for (int st = 0; st < 2; ++st) {
        const _Float16* kr = &ldsK[(st * 16 + lh) * kD];
        V16H a0, a1;  // A[m][k] = K[s0+16*st+m][32*dc + k], contiguous k = 16*cc + j
        a0.q[0] = *(const uint4*)(kr + cc * 16);
        a0.q[1] = *(const uint4*)(kr + 8 + cc * 16);
        a1.q[0] = *(const uint4*)(kr + 32 + cc * 16);
        a1.q[1] = *(const uint4*)(kr + 40 + cc * 16);
        v8f s_acc = __builtin_amdgcn_wmma_f32_16x16x32_f16(
            false, a0.v, false, bq[0].v, (short)0, zero8, false, false);
        s_acc = __builtin_amdgcn_wmma_f32_16x16x32_f16(
            false, a1.v, false, bq[1].v, (short)0, s_acc, false, false);
        sc[st] = s_acc;
      }

      // ---- mask (s <= t), scale by exp(-n[t]), pack to f16 dwords ----
      const bool full = (s0 + kChunkS - 1) <= t_base;
      unsigned pk[2][4];
#pragma unroll
      for (int st = 0; st < 2; ++st) {
        V8F x; x.v = sc[st];
#pragma unroll
        for (int r = 0; r < 8; ++r) {
          const int sg = s0 + st * 16 + r + 8 * cc;   // C layout: M = r + 8*cc
          x.f[r] = (full || sg <= t) ? x.f[r] * scale : 0.0f;
        }
#pragma unroll
        for (int k = 0; k < 4; ++k) {
          H2 pr;
          pr.h[0] = (_Float16)x.f[2 * k];
          pr.h[1] = (_Float16)x.f[2 * k + 1];
          pk[st][k] = pr.u;
        }
      }

      // ---- build P^T B-operand (32(s) x 16(t)) via cross-lane bpermute ----
      // B dword d (halves k=16*cc+2d..): source lane = lh (+16 for d>=4),
      // source tile = cc, source packed reg = d&3.
      const int idx_lo = lh * 4;
      const int idx_hi = idx_lo + 64;
      V16H bp;
#pragma unroll
      for (int k = 0; k < 4; ++k) {
        const int lo0 = __builtin_amdgcn_ds_bpermute(idx_lo, (int)pk[0][k]);
        const int lo1 = __builtin_amdgcn_ds_bpermute(idx_lo, (int)pk[1][k]);
        const int hi0 = __builtin_amdgcn_ds_bpermute(idx_hi, (int)pk[0][k]);
        const int hi1 = __builtin_amdgcn_ds_bpermute(idx_hi, (int)pk[1][k]);
        bp.u[k]     = cc ? (unsigned)lo1 : (unsigned)lo0;
        bp.u[4 + k] = cc ? (unsigned)hi1 : (unsigned)hi0;
      }

      // ---- O^T += V^T · P^T : 4 d-tiles, K = 32 keys ----
#pragma unroll
      for (int nt = 0; nt < 4; ++nt) {
        const _Float16* vr = &ldsV[(nt * 16 + lh) * kChunkS];
        V16H a;  // A[m][k] = V[s0+k][16*nt+m]
        a.q[0] = *(const uint4*)(vr + cc * 16);
        a.q[1] = *(const uint4*)(vr + 8 + cc * 16);
        acc[nt] = __builtin_amdgcn_wmma_f32_16x16x32_f16(
            false, a.v, false, bp.v, (short)0, acc[nt], false, false);
      }
    }
    __syncthreads();
  }

  // ---- write out[b][t][h][:] : contiguous d runs per lane ----
  float* po = out + (((size_t)(b * kS + t)) * kH + h) * kD;
#pragma unroll
  for (int nt = 0; nt < 4; ++nt) {
    V8F x; x.v = acc[nt];
    *(float4*)(po + nt * 16 + cc * 8)     = make_float4(x.f[0], x.f[1], x.f[2], x.f[3]);
    *(float4*)(po + nt * 16 + cc * 8 + 4) = make_float4(x.f[4], x.f[5], x.f[6], x.f[7]);
  }
}

extern "C" void kernel_launch(void* const* d_in, const int* in_sizes, int n_in,
                              void* d_out, int out_size, void* d_ws, size_t ws_size,
                              hipStream_t stream) {
  (void)in_sizes; (void)n_in; (void)out_size; (void)d_ws; (void)ws_size;
  const float* qk = (const float*)d_in[0];
  const float* v  = (const float*)d_in[1];
  const float* n  = (const float*)d_in[2];
  float* out = (float*)d_out;
  // grid = B*H*(S/128): blockIdx = b*256 + h*16 + qt
  dim3 grid(kB * kH * (kS / kBlockQ));
  attn_fwd<<<grid, 256, 0, stream>>>(qk, v, n, out);
}
